// Model_62577673503271
// MI455X (gfx1250) — matrix operand order repaired and checked by reading, not compiled, not run on hardware
//
#include <hip/hip_runtime.h>
#include <cstdint>

typedef __attribute__((ext_vector_type(16))) __bf16 v16bf;
typedef __attribute__((ext_vector_type(8)))  float  v8f;

#define DEVINL __device__ __forceinline__

constexpr int Bq   = 1024;
constexpr int NC   = 100000;
constexpr int FIN  = 100;
constexpr int FP   = 128;     // padded feature dim
constexpr int Dm   = 256;
constexpr int DBm  = 512;
constexpr int Sctx = 96;
constexpr int NSEG = 8;
constexpr float NEGF = -1e30f;

DEVINL v8f vzero8() { v8f z = {0.f,0.f,0.f,0.f,0.f,0.f,0.f,0.f}; return z; }

DEVINL v8f wmma_bf16(v16bf a, v16bf b, v8f c) {
  return __builtin_amdgcn_wmma_f32_16x16x32_bf16(false, a, false, b, (short)0, c, false, false);
}

// A fragment: 16x32 (MxK), row-major source, row stride ld (elements).
// lanes 0-15: row m, K = k..k+15; lanes 16-31: K = k+16..k+31
DEVINL v16bf load_frag_a(const __bf16* base, int ld) {
  int lane = threadIdx.x & 31;
  int m  = lane & 15;
  int kb = (lane >> 4) ? 16 : 0;
  const __bf16* p = base + (size_t)m * ld + kb;
  v16bf a;
#pragma unroll
  for (int j = 0; j < 8; ++j) a[j] = p[j];
#pragma unroll
  for (int j = 0; j < 8; ++j) a[8+j] = p[8+j];
  return a;
}

// B fragment: 32x16 (KxN); source stored N-major ([N,K] row-major), stride ld.
// lane n holds col n: lanes 0-15 K=k..k+15, lanes 16-31 K=k+16..k+31
DEVINL v16bf load_frag_b(const __bf16* base, int ld) {
  int lane = threadIdx.x & 31;
  int n  = lane & 15;
  int kb = (lane >> 4) ? 16 : 0;
  const __bf16* p = base + (size_t)n * ld + kb;
  v16bf b;
#pragma unroll
  for (int j = 0; j < 16; ++j) b[j] = p[j];
  return b;
}

// ---------------- prep: f32 -> bf16 conversions ----------------
__global__ void cvt_pad_rows(const float* __restrict__ src, __bf16* __restrict__ dst,
                             int rows, int C, int Cp) {
  size_t i = (size_t)blockIdx.x * 256 + threadIdx.x;
  size_t total = (size_t)rows * Cp;
  if (i >= total) return;
  int r = (int)(i / Cp), c = (int)(i % Cp);
  dst[i] = (c < C) ? (__bf16)src[(size_t)r * C + c] : (__bf16)0.f;
}

// dst[n*Kp+k] = src[k*Nn+n]  (weight [K,Nn] -> transposed-padded [Nn,Kp] bf16)
__global__ void cvt_transpose(const float* __restrict__ src, __bf16* __restrict__ dst,
                              int K, int Nn, int Kp) {
  size_t i = (size_t)blockIdx.x * 256 + threadIdx.x;
  size_t total = (size_t)Nn * Kp;
  if (i >= total) return;
  int n = (int)(i / Kp), k = (int)(i % Kp);
  dst[i] = (k < K) ? (__bf16)src[(size_t)k * Nn + n] : (__bf16)0.f;
}

// ---------------- encoder: 16 rows per WG, full MLP chain via WMMA ----------------
__global__ __launch_bounds__(256) void encode_kernel(
    const __bf16* __restrict__ Xp,
    const __bf16* __restrict__ WlinT, const float* __restrict__ blin,
    const __bf16* __restrict__ W01T,  const float* __restrict__ b01,
    const __bf16* __restrict__ W02T,  const float* __restrict__ b02,
    const float* __restrict__ g_mix,  const float* __restrict__ be_mix,
    const __bf16* __restrict__ WKT,   const float* __restrict__ bK,
    __bf16* __restrict__ out_kb, float* __restrict__ out_csq,
    float* __restrict__ out_h)
{
  __shared__ __bf16 sx[16][FP];
  __shared__ float  sh[16][Dm];
  __shared__ __bf16 shb[16][Dm];
  __shared__ __bf16 st[16][DBm];
  __shared__ __bf16 sln[16][Dm];
  __shared__ float  scsq[16];

  const int tid = threadIdx.x, w = tid >> 5, lane = tid & 31;
  const int n16 = lane & 15, mh = (lane >> 4) * 8;
  const int row0 = blockIdx.x * 16;

#pragma unroll
  for (int j = 0; j < 8; ++j) {
    int e = tid * 8 + j;
    ((__bf16*)sx)[e] = Xp[(size_t)row0 * FP + e];
  }
  if (tid < 16) scsq[tid] = 0.f;
  __syncthreads();

  // Stage A: h = x @ Wlin + blin
  {
    v16bf aF[4];
#pragma unroll
    for (int kk = 0; kk < 4; ++kk) aF[kk] = load_frag_a(&sx[0][kk*32], FP);
#pragma unroll
    for (int i = 0; i < 2; ++i) {
      int nt = w * 2 + i;
      v8f acc = vzero8();
#pragma unroll
      for (int kk = 0; kk < 4; ++kk)
        acc = wmma_bf16(aF[kk], load_frag_b(WlinT + (size_t)nt*16*FP + kk*32, FP), acc);
      int n = nt*16 + n16;
      float bias = blin[n];
#pragma unroll
      for (int r = 0; r < 8; ++r) {
        float v = acc[r] + bias;
        sh[mh+r][n]  = v;
        shb[mh+r][n] = (__bf16)v;
      }
    }
  }
  __syncthreads();

  // Stage B: t = relu(h @ W01 + b01)
  {
    v16bf aF[8];
#pragma unroll
    for (int kk = 0; kk < 8; ++kk) aF[kk] = load_frag_a(&shb[0][kk*32], Dm);
#pragma unroll
    for (int i = 0; i < 4; ++i) {
      int nt = w * 4 + i;
      v8f acc = vzero8();
#pragma unroll
      for (int kk = 0; kk < 8; ++kk)
        acc = wmma_bf16(aF[kk], load_frag_b(W01T + (size_t)nt*16*Dm + kk*32, Dm), acc);
      int n = nt*16 + n16;
      float bias = b01[n];
#pragma unroll
      for (int r = 0; r < 8; ++r)
        st[mh+r][n] = (__bf16)fmaxf(acc[r] + bias, 0.f);
    }
  }
  __syncthreads();

  // Stage C: h2 = h + t @ W02 + b02
  {
#pragma unroll
    for (int i = 0; i < 2; ++i) {
      int nt = w * 2 + i;
      v8f acc = vzero8();
#pragma unroll
      for (int kk = 0; kk < 16; ++kk)
        acc = wmma_bf16(load_frag_a(&st[0][kk*32], DBm),
                        load_frag_b(W02T + (size_t)nt*16*DBm + kk*32, DBm), acc);
      int n = nt*16 + n16;
      float bias = b02[n];
#pragma unroll
      for (int r = 0; r < 8; ++r)
        sh[mh+r][n] += acc[r] + bias;
    }
  }
  __syncthreads();

  // LayerNorm -> sln (bf16); wave w handles rows 2w, 2w+1
#pragma unroll
  for (int rr = 0; rr < 2; ++rr) {
    int m = w*2 + rr;
    float s = 0.f;
    for (int j = lane; j < Dm; j += 32) s += sh[m][j];
#pragma unroll
    for (int off = 16; off >= 1; off >>= 1) s += __shfl_xor(s, off);
    float mean = s * (1.f/Dm);
    float vs = 0.f;
    for (int j = lane; j < Dm; j += 32) { float d = sh[m][j]-mean; vs += d*d; }
#pragma unroll
    for (int off = 16; off >= 1; off >>= 1) vs += __shfl_xor(vs, off);
    float rstd = rsqrtf(vs * (1.f/Dm) + 1e-5f);
    for (int j = lane; j < Dm; j += 32)
      sln[m][j] = (__bf16)((sh[m][j]-mean)*rstd*g_mix[j] + be_mix[j]);
  }
  __syncthreads();

  // Stage D: k = ln @ WK + bK
  {
    v16bf aF[8];
#pragma unroll
    for (int kk = 0; kk < 8; ++kk) aF[kk] = load_frag_a(&sln[0][kk*32], Dm);
#pragma unroll
    for (int i = 0; i < 2; ++i) {
      int nt = w * 2 + i;
      v8f acc = vzero8();
#pragma unroll
      for (int kk = 0; kk < 8; ++kk)
        acc = wmma_bf16(aF[kk], load_frag_b(WKT + (size_t)nt*16*Dm + kk*32, Dm), acc);
      int n = nt*16 + n16;
      float bias = bK[n];
#pragma unroll
      for (int r = 0; r < 8; ++r) {
        float v = acc[r] + bias;
        out_kb[(size_t)(row0+mh+r)*Dm + n] = (__bf16)v;
        if (out_csq) atomicAdd(&scsq[mh+r], v*v);
      }
    }
  }
  __syncthreads();
  if (out_csq && tid < 16) out_csq[row0 + tid] = scsq[tid];
  if (out_h)
    for (int e = tid; e < 16*Dm; e += 256)
      out_h[(size_t)row0*Dm + e] = ((float*)sh)[e];
}

// ---------------- top-k machinery: 96-entry list distributed over one wave ----------------
DEVINL void list_insert(float v, int id,
                        float (&lv)[3], int (&li)[3],
                        float &wmin, int &wpos, int lane)
{
  bool want = v > wmin;
  unsigned long long m = __ballot(want);
  while (m) {
    int src = __ffsll(m) - 1;
    float vv = __shfl(v, src);
    int   ii = __shfl(id, src);
    if (vv > wmin) {
      if (lane == (wpos & 31)) {
        int sl = wpos >> 5;
        if (sl == 0)      { lv[0] = vv; li[0] = ii; }
        else if (sl == 1) { lv[1] = vv; li[1] = ii; }
        else              { lv[2] = vv; li[2] = ii; }
      }
      float bm = lv[0]; int bp = lane;
      if (lv[1] < bm) { bm = lv[1]; bp = lane + 32; }
      if (lv[2] < bm) { bm = lv[2]; bp = lane + 64; }
#pragma unroll
      for (int off = 16; off >= 1; off >>= 1) {
        float om = __shfl_xor(bm, off);
        int   op = __shfl_xor(bp, off);
        if (om < bm || (om == bm && op < bp)) { bm = om; bp = op; }
      }
      wmin = bm; wpos = bp;
    }
    m &= (m - 1);
  }
}

// 16 queries per WG, one N-segment per blockIdx.y; scores s = 2*k.ck - |ck|^2
__global__ __launch_bounds__(256) void knn_partial(
    const __bf16* __restrict__ q_kb, const __bf16* __restrict__ cand_kb,
    const float* __restrict__ csq,
    float* __restrict__ psv, int* __restrict__ psi)
{
  __shared__ float dmat[16][132];
  const int tid = threadIdx.x, w = tid >> 5, lane = tid & 31;
  const int n16 = lane & 15, mh = (lane >> 4) * 8;
  const int qt = blockIdx.x, seg = blockIdx.y;
  const int CHT = (NC + 127) / 128;           // 782 chunks of 128 candidates
  const int CPS = (CHT + NSEG - 1) / NSEG;    // 98 per segment
  const int c_begin = seg * CPS;
  const int c_end   = min(CHT, c_begin + CPS);

  v16bf qa[8];
#pragma unroll
  for (int kk = 0; kk < 8; ++kk)
    qa[kk] = load_frag_a(q_kb + (size_t)qt*16*Dm + kk*32, Dm);

  float lv[2][3] = {{NEGF,NEGF,NEGF},{NEGF,NEGF,NEGF}};
  int   li[2][3] = {{0,0,0},{0,0,0}};
  float wmin[2] = {NEGF, NEGF};
  int   wpos[2] = {0, 0};

  for (int ch = c_begin; ch < c_end; ++ch) {
    int c0 = ch*128 + w*16;
    int crow = min(c0 + n16, NC - 1);
    const __bf16* bb = cand_kb + (size_t)crow * Dm + ((lane >> 4) ? 16 : 0);
    {
      int nrow = min(c0 + 128 + n16, NC - 1);        // stream hint for next chunk
      __builtin_prefetch(cand_kb + (size_t)nrow * Dm, 0, 1);
    }
    v8f acc = vzero8();
#pragma unroll
    for (int kk = 0; kk < 8; ++kk) {
      v16bf b;
#pragma unroll
      for (int j = 0; j < 16; ++j) b[j] = bb[kk*32 + j];
      acc = wmma_bf16(qa[kk], b, acc);
    }
    float cs = csq[crow];
    bool valid = (c0 + n16) < NC;
#pragma unroll
    for (int r = 0; r < 8; ++r)
      dmat[mh+r][w*16 + n16] = valid ? (2.f*acc[r] - cs) : NEGF;
    __syncthreads();
    int cbase = ch * 128;
#pragma unroll
    for (int qq = 0; qq < 2; ++qq) {
      int q = w*2 + qq;
#pragma unroll
      for (int slot = 0; slot < 4; ++slot) {
        int col = lane + slot*32;
        list_insert(dmat[q][col], cbase + col, lv[qq], li[qq], wmin[qq], wpos[qq], lane);
      }
    }
    __syncthreads();
  }
#pragma unroll
  for (int qq = 0; qq < 2; ++qq) {
    size_t base = ((size_t)(qt*16 + w*2 + qq) * NSEG + seg) * Sctx;
#pragma unroll
    for (int e = 0; e < 3; ++e) {
      psv[base + lane + e*32] = lv[qq][e];
      psi[base + lane + e*32] = li[qq][e];
    }
  }
}

// merge 8 partial lists -> top-96 sorted desc, softmax probs, ctx indices
__global__ __launch_bounds__(256) void knn_merge(
    const float* __restrict__ psv, const int* __restrict__ psi,
    const int* __restrict__ cand_idx,
    int* __restrict__ out_ctx, float* __restrict__ out_probs,
    int* __restrict__ ctx_pos)
{
  __shared__ float ssv[8][96];  __shared__ int ssi[8][96];
  __shared__ float ssv2[8][96]; __shared__ int ssi2[8][96];
  const int tid = threadIdx.x, w = tid >> 5, lane = tid & 31;
  const int q = blockIdx.x * 8 + w;

  float lv[3] = {NEGF, NEGF, NEGF};
  int   li[3] = {0, 0, 0};
  float wmin = NEGF; int wpos = 0;

  size_t base = (size_t)q * NSEG * Sctx;     // 768 entries
  for (int t0 = 0; t0 < NSEG*Sctx; t0 += 32)
    list_insert(psv[base + t0 + lane], psi[base + t0 + lane], lv, li, wmin, wpos, lane);

#pragma unroll
  for (int e = 0; e < 3; ++e) { ssv[w][lane + e*32] = lv[e]; ssi[w][lane + e*32] = li[e]; }
  __syncthreads();
#pragma unroll
  for (int e = 0; e < 3; ++e) {
    int i = lane + e*32;
    float si = ssv[w][i]; int ii = ssi[w][i]; int rank = 0;
    for (int j = 0; j < 96; ++j) {
      float sj = ssv[w][j]; int ij = ssi[w][j];
      rank += (sj > si) || (sj == si && ij < ii);
    }
    ssv2[w][rank] = si; ssi2[w][rank] = ii;
  }
  __syncthreads();
  float smax = ssv2[w][0];
  float pe[3]; float sum = 0.f;
#pragma unroll
  for (int e = 0; e < 3; ++e) { pe[e] = __expf(ssv2[w][lane + e*32] - smax); sum += pe[e]; }
#pragma unroll
  for (int off = 16; off >= 1; off >>= 1) sum += __shfl_xor(sum, off);
#pragma unroll
  for (int e = 0; e < 3; ++e) {
    int i = lane + e*32;
    int pos = ssi2[w][i];
    out_probs[(size_t)q*Sctx + i] = pe[e] / sum;
    ctx_pos[(size_t)q*Sctx + i]   = pos;
    out_ctx[(size_t)q*Sctx + i]   = cand_idx[pos];
  }
}

// values MLP + prob-weighted reduction; grid (B, 2) -> 48 context rows per WG
__global__ __launch_bounds__(256) void values_kernel(
    const int* __restrict__ ctx_pos, const float* __restrict__ probs,
    const float* __restrict__ cand_y,
    const __bf16* __restrict__ q_kb, const __bf16* __restrict__ cand_kb,
    const __bf16* __restrict__ WT1T, const float* __restrict__ bT1,
    const __bf16* __restrict__ WT2T,
    float* __restrict__ ctx_x, float* __restrict__ wyb)
{
  __shared__ __bf16 sdiff[48][Dm];
  __shared__ __bf16 st2[48][128];
  __shared__ float sp[48];
  __shared__ int   sidx[48];
  __shared__ float sacc[Dm];
  __shared__ float swy;
  const int tid = threadIdx.x, w = tid >> 5, lane = tid & 31;
  const int n16 = lane & 15, mh = (lane >> 4) * 8;
  const int b = blockIdx.x, half = blockIdx.y;
  const int s0 = half * 48;

  if (tid < 48) {
    sp[tid]   = probs[(size_t)b*Sctx + s0 + tid];
    sidx[tid] = ctx_pos[(size_t)b*Sctx + s0 + tid];
  }
  if (tid == 0) swy = 0.f;
  if (tid < Dm) sacc[tid] = 0.f;
  __syncthreads();
  if (tid < 48) atomicAdd(&swy, sp[tid] * cand_y[sidx[tid]]);
  for (int e = tid; e < 48*Dm; e += 256) {
    int s = e >> 8, c = e & 255;
    float qv = (float)q_kb[(size_t)b*Dm + c];
    float cv = (float)cand_kb[(size_t)sidx[s]*Dm + c];
    sdiff[s][c] = (__bf16)(qv - cv);
  }
  __syncthreads();

  v8f accT[6];
#pragma unroll
  for (int i = 0; i < 6; ++i) accT[i] = vzero8();

  for (int q4 = 0; q4 < 4; ++q4) {                  // 128-wide chunk of DB=512
#pragma unroll
    for (int i = 0; i < 3; ++i) {                   // GEMM1: 24 tiles / 8 waves
      int t = w*3 + i;
      int mt = t % 3, ntl = t / 3;
      int ng = q4*128 + ntl*16;
      v8f acc = vzero8();
#pragma unroll
      for (int kk = 0; kk < 8; ++kk)
        acc = wmma_bf16(load_frag_a(&sdiff[mt*16][kk*32], Dm),
                        load_frag_b(WT1T + (size_t)ng*Dm + kk*32, Dm), acc);
      float bias = bT1[ng + n16];
#pragma unroll
      for (int r = 0; r < 8; ++r)
        st2[mt*16+mh+r][ntl*16+n16] = (__bf16)fmaxf(acc[r] + bias, 0.f);
    }
    __syncthreads();
#pragma unroll
    for (int i = 0; i < 6; ++i) {                   // GEMM2 partial-k: 48 tiles / 8 waves
      int t = w*6 + i;
      int mt = t % 3, nt = t / 3;
#pragma unroll
      for (int kk = 0; kk < 4; ++kk)
        accT[i] = wmma_bf16(load_frag_a(&st2[mt*16][kk*32], 128),
                            load_frag_b(WT2T + (size_t)nt*16*DBm + q4*128 + kk*32, DBm),
                            accT[i]);
    }
    __syncthreads();
  }
#pragma unroll
  for (int i = 0; i < 6; ++i) {
    int t = w*6 + i;
    int mt = t % 3, nt = t / 3;
#pragma unroll
    for (int r = 0; r < 8; ++r) {
      float p = sp[mt*16 + mh + r];
      atomicAdd(&sacc[nt*16 + n16], p * accT[i][r]);
    }
  }
  __syncthreads();
  if (tid < Dm) atomicAdd(&ctx_x[(size_t)b*Dm + tid], sacc[tid]);
  if (tid == 0) atomicAdd(&wyb[b], swy);
}

// final predictor block + head (tiny, VALU)
__global__ __launch_bounds__(256) void final_kernel(
    const float* __restrict__ x_h, const float* __restrict__ ctx_x,
    const float* __restrict__ wyb, const float* __restrict__ Wle, const float* __restrict__ ble,
    const float* __restrict__ g1, const float* __restrict__ be1,
    const float* __restrict__ W11, const float* __restrict__ b11,
    const float* __restrict__ W12, const float* __restrict__ b12,
    const float* __restrict__ gh, const float* __restrict__ beh,
    const float* __restrict__ Wh, const float* __restrict__ bh,
    float* __restrict__ y_pred)
{
  __shared__ float sln[Dm];
  __shared__ float stv[DBm];
  __shared__ float red[8];
  const int tid = threadIdx.x, w = tid >> 5, lane = tid & 31;
  const int b = blockIdx.x;

  auto blocksum = [&](float v) -> float {
#pragma unroll
    for (int off = 16; off >= 1; off >>= 1) v += __shfl_xor(v, off);
    __syncthreads();
    if (lane == 0) red[w] = v;
    __syncthreads();
    float s = 0.f;
#pragma unroll
    for (int i = 0; i < 8; ++i) s += red[i];
    return s;
  };

  float h = x_h[(size_t)b*Dm + tid] + ctx_x[(size_t)b*Dm + tid]
          + wyb[b]*Wle[tid] + ble[tid];

  float mean = blocksum(h) * (1.f/Dm);
  float d = h - mean;
  float var = blocksum(d*d) * (1.f/Dm);
  float rstd = rsqrtf(var + 1e-5f);
  sln[tid] = (h - mean)*rstd*g1[tid] + be1[tid];
  __syncthreads();

#pragma unroll
  for (int jj = 0; jj < 2; ++jj) {
    int j = tid + jj*256;
    float acc = b11[j];
    for (int c = 0; c < Dm; ++c) acc += sln[c] * W11[(size_t)c*DBm + j];
    stv[j] = fmaxf(acc, 0.f);
  }
  __syncthreads();
  float acc2 = b12[tid];
  for (int j = 0; j < DBm; ++j) acc2 += stv[j] * W12[(size_t)j*Dm + tid];
  float h2 = h + acc2;

  float mean2 = blocksum(h2) * (1.f/Dm);
  float d2 = h2 - mean2;
  float var2 = blocksum(d2*d2) * (1.f/Dm);
  float rstd2 = rsqrtf(var2 + 1e-5f);
  float r = fmaxf((h2 - mean2)*rstd2*gh[tid] + beh[tid], 0.f);
  float tot = blocksum(r * Wh[tid]);
  if (tid == 0) y_pred[b] = tot + bh[0];
}

// ---------------- host ----------------
extern "C" void kernel_launch(void* const* d_in, const int* in_sizes, int n_in,
                              void* d_out, int out_size, void* d_ws, size_t ws_size,
                              hipStream_t stream)
{
  (void)in_sizes; (void)n_in; (void)out_size; (void)ws_size;
  const float* x_num    = (const float*)d_in[0];
  const float* cand_x   = (const float*)d_in[1];
  const float* cand_y   = (const float*)d_in[2];
  const int*   cand_idx = (const int*)d_in[3];
  const float* Wlin = (const float*)d_in[4];
  const float* blin = (const float*)d_in[5];
  const float* W01  = (const float*)d_in[6];
  const float* b01  = (const float*)d_in[7];
  const float* W02  = (const float*)d_in[8];
  const float* b02  = (const float*)d_in[9];
  const float* g_mix  = (const float*)d_in[10];
  const float* be_mix = (const float*)d_in[11];
  const float* WK   = (const float*)d_in[12];
  const float* bK   = (const float*)d_in[13];
  const float* Wle  = (const float*)d_in[14];
  const float* ble  = (const float*)d_in[15];
  const float* WT1  = (const float*)d_in[16];
  const float* bT1  = (const float*)d_in[17];
  const float* WT2  = (const float*)d_in[18];
  const float* g1   = (const float*)d_in[19];
  const float* be1  = (const float*)d_in[20];
  const float* W11  = (const float*)d_in[21];
  const float* b11  = (const float*)d_in[22];
  const float* W12  = (const float*)d_in[23];
  const float* b12  = (const float*)d_in[24];
  const float* gh   = (const float*)d_in[25];
  const float* beh  = (const float*)d_in[26];
  const float* Wh   = (const float*)d_in[27];
  const float* bh   = (const float*)d_in[28];

  char* ws = (char*)d_ws;
  size_t off = 0;
  auto take = [&](size_t bytes) -> char* {
    char* p = ws + off;
    off += (bytes + 255) & ~(size_t)255;
    return p;
  };
  __bf16* cand_xb  = (__bf16*)take((size_t)NC*FP*2);
  __bf16* x_b      = (__bf16*)take((size_t)Bq*FP*2);
  __bf16* WlinT    = (__bf16*)take((size_t)Dm*FP*2);
  __bf16* W01T     = (__bf16*)take((size_t)DBm*Dm*2);
  __bf16* W02T     = (__bf16*)take((size_t)Dm*DBm*2);
  __bf16* WKT      = (__bf16*)take((size_t)Dm*Dm*2);
  __bf16* WT1T     = (__bf16*)take((size_t)DBm*Dm*2);
  __bf16* WT2T     = (__bf16*)take((size_t)Dm*DBm*2);
  __bf16* cand_kb  = (__bf16*)take((size_t)NC*Dm*2);
  float*  cand_csq = (float*)take((size_t)NC*4);
  __bf16* q_kb     = (__bf16*)take((size_t)Bq*Dm*2);
  float*  x_h      = (float*)take((size_t)Bq*Dm*4);
  float*  ctx_x    = (float*)take((size_t)Bq*Dm*4);
  float*  wyb      = (float*)take((size_t)Bq*4);
  float*  psv      = (float*)take((size_t)Bq*NSEG*Sctx*4);
  int*    psi      = (int*)take((size_t)Bq*NSEG*Sctx*4);
  int*    ctx_pos  = (int*)take((size_t)Bq*Sctx*4);

  float* y_pred    = (float*)d_out;
  int*   out_ctx   = (int*)((float*)d_out + Bq);
  float* out_probs = (float*)d_out + Bq + (size_t)Bq*Sctx;

  cvt_pad_rows<<<(int)(((size_t)NC*FP + 255)/256), 256, 0, stream>>>(cand_x, cand_xb, NC, FIN, FP);
  cvt_pad_rows<<<(Bq*FP + 255)/256, 256, 0, stream>>>(x_num, x_b, Bq, FIN, FP);
  cvt_transpose<<<(Dm*FP + 255)/256, 256, 0, stream>>>(Wlin, WlinT, FIN, Dm, FP);
  cvt_transpose<<<(DBm*Dm + 255)/256, 256, 0, stream>>>(W01, W01T, Dm, DBm, Dm);
  cvt_transpose<<<(Dm*DBm + 255)/256, 256, 0, stream>>>(W02, W02T, DBm, Dm, DBm);
  cvt_transpose<<<(Dm*Dm + 255)/256, 256, 0, stream>>>(WK, WKT, Dm, Dm, Dm);
  cvt_transpose<<<(DBm*Dm + 255)/256, 256, 0, stream>>>(WT1, WT1T, Dm, DBm, Dm);
  cvt_transpose<<<(Dm*DBm + 255)/256, 256, 0, stream>>>(WT2, WT2T, DBm, Dm, DBm);

  encode_kernel<<<NC/16, 256, 0, stream>>>(cand_xb, WlinT, blin, W01T, b01, W02T, b02,
                                           g_mix, be_mix, WKT, bK,
                                           cand_kb, cand_csq, nullptr);
  encode_kernel<<<Bq/16, 256, 0, stream>>>(x_b, WlinT, blin, W01T, b01, W02T, b02,
                                           g_mix, be_mix, WKT, bK,
                                           q_kb, nullptr, x_h);

  hipMemsetAsync(ctx_x, 0, (size_t)Bq*Dm*4, stream);
  hipMemsetAsync(wyb, 0, (size_t)Bq*4, stream);

  knn_partial<<<dim3(Bq/16, NSEG), 256, 0, stream>>>(q_kb, cand_kb, cand_csq, psv, psi);
  knn_merge<<<Bq/8, 256, 0, stream>>>(psv, psi, cand_idx, out_ctx, out_probs, ctx_pos);
  values_kernel<<<dim3(Bq, 2), 256, 0, stream>>>(ctx_pos, out_probs, cand_y, q_kb, cand_kb,
                                                 WT1T, bT1, WT2T, ctx_x, wyb);
  final_kernel<<<Bq, 256, 0, stream>>>(x_h, ctx_x, wyb, Wle, ble,
                                       g1, be1, W11, b11, W12, b12,
                                       gh, beh, Wh, bh, y_pred);
}
